// MultiHeadAttentionOld_88003879895175
// MI455X (gfx1250) — matrix operand order repaired and checked by reading, not compiled, not run on hardware
//
#include <hip/hip_runtime.h>

typedef __attribute__((ext_vector_type(16))) __bf16 v16bf;
typedef __attribute__((ext_vector_type(8)))  __bf16 v8bf;
typedef __attribute__((ext_vector_type(8)))  float  v8f;

#define BB   4
#define TT   2048
#define CC   2048
#define NH   16
#define NKV  4
#define HD   128
#define KVC  (NKV*HD)   // 512
#define MT   (BB*TT)    // 8192
#define KB   64         // flash-attention key block

// ---------------------------------------------------------------- helpers
__device__ __forceinline__ v8f zero_v8f() {
    v8f z;
#pragma unroll
    for (int i = 0; i < 8; ++i) z[i] = 0.0f;
    return z;
}

__device__ __forceinline__ v16bf combine16(v8bf lo, v8bf hi) {
    v16bf r;
#pragma unroll
    for (int i = 0; i < 8; ++i) { r[i] = lo[i]; r[8 + i] = hi[i]; }
    return r;
}

// ---------------------------------------------------------------- fp32 -> bf16
__global__ __launch_bounds__(256) void cvt_f32_bf16(const float* __restrict__ src,
                                                    __bf16* __restrict__ dst, long n) {
    long i = (long)blockIdx.x * blockDim.x + threadIdx.x;
    if (i < n) dst[i] = (__bf16)src[i];
}

// ---------------------------------------------------------------- GEMM: O[m,n] = sum_k A[m,k]*W[n,k]
// A: MxK bf16 row-major. W: NxK bf16 row-major (weight layout). O: MxN fp32.
// One wave per 32(M) x 64(N) tile: each B fragment feeds 2 WMMAs. K multiple of 32.
__global__ __launch_bounds__(32) void gemm_bf16_nt(const __bf16* __restrict__ A,
                                                   const __bf16* __restrict__ W,
                                                   float* __restrict__ O,
                                                   int M, int N, int K) {
    const int lane = threadIdx.x;
    const int half = lane >> 4;
    const int lm   = lane & 15;
    const int m0 = blockIdx.x * 32;
    const int n0 = blockIdx.y * 64;

    const __bf16* arow0 = A + (long)(m0 + lm) * K;       // A rows for M tile 0
    const __bf16* arow1 = A + (long)(m0 + 16 + lm) * K;  // A rows for M tile 1
    const __bf16* wrow[4];
#pragma unroll
    for (int nt = 0; nt < 4; ++nt)                       // B: lane holds column N=n0+nt*16+lm
        wrow[nt] = W + (long)(n0 + nt * 16 + lm) * K;

    v8f acc[8];
#pragma unroll
    for (int i = 0; i < 8; ++i) acc[i] = zero_v8f();

    for (int kk = 0; kk < K; kk += 32) {
        // A 16x32 fragments: lanes 0-15 hold K {0..7,16..23}, lanes 16-31 hold K {8..15,24..31}
        v16bf a0 = combine16(*(const v8bf*)(arow0 + kk + half * 8),
                             *(const v8bf*)(arow0 + kk + 16 + half * 8));
        v16bf a1 = combine16(*(const v8bf*)(arow1 + kk + half * 8),
                             *(const v8bf*)(arow1 + kk + 16 + half * 8));
#pragma unroll
        for (int nt = 0; nt < 4; ++nt) {
            v16bf b = combine16(*(const v8bf*)(wrow[nt] + kk + half * 8),    // same K order as A
                                *(const v8bf*)(wrow[nt] + kk + 16 + half * 8));
            acc[nt]     = __builtin_amdgcn_wmma_f32_16x16x32_bf16(
                false, a0, false, b, (short)0, acc[nt],     false, false);
            acc[4 + nt] = __builtin_amdgcn_wmma_f32_16x16x32_bf16(
                false, a1, false, b, (short)0, acc[4 + nt], false, false);
        }
    }
    // C layout: VGPR r -> row (mt*16 + r + 8*half), column n0 + nt*16 + (lane&15)
#pragma unroll
    for (int mt = 0; mt < 2; ++mt)
#pragma unroll
        for (int nt = 0; nt < 4; ++nt)
#pragma unroll
            for (int r = 0; r < 8; ++r)
                O[(long)(m0 + mt * 16 + r + 8 * half) * N + n0 + nt * 16 + lm] =
                    acc[mt * 4 + nt][r];
}

// ---------------------------------------------------------------- RoPE Q: fp32 [B*T, C] -> bf16 [B,NH,T,HD]
__global__ __launch_bounds__(256) void rope_q_kernel(const float* __restrict__ q,
                                                     const float* __restrict__ freqs,
                                                     __bf16* __restrict__ qr) {
    long i = (long)blockIdx.x * blockDim.x + threadIdx.x;
    long total = (long)BB * NH * TT * (HD / 2);
    if (i >= total) return;
    int j = (int)(i & 63); long r = i >> 6;
    int t = (int)(r & (TT - 1)); r >>= 11;
    int h = (int)(r & (NH - 1)); int b = (int)(r >> 4);
    float ang = freqs[t * (HD / 2) + j];
    float c = __cosf(ang), s = __sinf(ang);
    const float* src = q + (long)(b * TT + t) * CC + h * HD + 2 * j;
    float e = src[0], o = src[1];
    __bf16* dst = qr + (((long)(b * NH + h) * TT + t) * HD) + 2 * j;
    dst[0] = (__bf16)(e * c - o * s);
    dst[1] = (__bf16)(e * s + o * c);
}

// ---------------------------------------------------------------- RoPE K: fp32 [B*T, KVC] -> bf16 [B,NKV,T,HD]
__global__ __launch_bounds__(256) void rope_k_kernel(const float* __restrict__ k,
                                                     const float* __restrict__ freqs,
                                                     __bf16* __restrict__ kr) {
    long i = (long)blockIdx.x * blockDim.x + threadIdx.x;
    long total = (long)BB * NKV * TT * (HD / 2);
    if (i >= total) return;
    int j = (int)(i & 63); long r = i >> 6;
    int t = (int)(r & (TT - 1)); r >>= 11;
    int kv = (int)(r & (NKV - 1)); int b = (int)(r >> 2);
    float ang = freqs[t * (HD / 2) + j];
    float c = __cosf(ang), s = __sinf(ang);
    const float* src = k + (long)(b * TT + t) * KVC + kv * HD + 2 * j;
    float e = src[0], o = src[1];
    __bf16* dst = kr + (((long)(b * NKV + kv) * TT + t) * HD) + 2 * j;
    dst[0] = (__bf16)(e * c - o * s);
    dst[1] = (__bf16)(e * s + o * c);
}

// ---------------------------------------------------------------- V pack: fp32 [B*T, KVC] -> bf16 [B,NKV,HD,T] (transposed)
__global__ __launch_bounds__(256) void pack_vT_kernel(const float* __restrict__ v,
                                                      __bf16* __restrict__ vT) {
    long i = (long)blockIdx.x * blockDim.x + threadIdx.x;
    long total = (long)BB * NKV * TT * HD;
    if (i >= total) return;
    int d = (int)(i & (HD - 1)); long r = i >> 7;
    int t = (int)(r & (TT - 1)); r >>= 11;
    int kv = (int)(r & (NKV - 1)); int b = (int)(r >> 2);
    vT[(((long)(b * NKV + kv) * HD + d) * TT) + t] =
        (__bf16)v[(long)(b * TT + t) * KVC + kv * HD + d];
}

// ---------------------------------------------------------------- Flash attention (online softmax)
// One wave per 16-query tile of one (b,h). Key blocks of 64 (softmax bookkeeping
// amortized over 32 WMMAs per block).
// qr: [B,NH,T,HD], kr: [B,NKV,T,HD], vT: [B,NKV,HD,T], ctx bf16: [B,T,C] (col = h*HD+d)
__global__ __launch_bounds__(32) void flash_attn_kernel(const __bf16* __restrict__ qr,
                                                        const __bf16* __restrict__ kr,
                                                        const __bf16* __restrict__ vT,
                                                        const int* __restrict__ amask,
                                                        __bf16* __restrict__ ctx) {
    __shared__ __attribute__((aligned(32))) __bf16 ldsP[16 * KB];  // 2 KB

    const int lane = threadIdx.x;
    const int half = lane >> 4;
    const int lm   = lane & 15;
    const int m0 = blockIdx.x * 16;
    const int h  = blockIdx.y;
    const int b  = blockIdx.z;
    const int kv = h >> 2;                     // NH/NKV = 4
    const float scale = 0.08838834764831845f;  // 1/sqrt(128)

    // Q fragments: 16 x 128 -> four 16x32 A-fragments, held for the whole key loop
    const __bf16* qbase = qr + ((long)(b * NH + h) * TT + m0 + lm) * HD;
    v16bf qf[4];
#pragma unroll
    for (int c = 0; c < 4; ++c)
        qf[c] = combine16(*(const v8bf*)(qbase + c * 32 + half * 8),
                          *(const v8bf*)(qbase + c * 32 + 16 + half * 8));

    v8f accO[8];
#pragma unroll
    for (int nc = 0; nc < 8; ++nc) accO[nc] = zero_v8f();
    float mrow[8], lrow[8];
#pragma unroll
    for (int r = 0; r < 8; ++r) { mrow[r] = -1e30f; lrow[r] = 0.0f; }

    const __bf16* kbase = kr + (long)(b * NKV + kv) * TT * HD;
    const __bf16* vbase = vT + (long)(b * NKV + kv) * HD * TT;
    const int*    am    = amask + (long)b * TT;

    const int nkb = (m0 + 16 + KB - 1) >> 6;   // causal: keys <= m0+15
    for (int kb = 0; kb < nkb; ++kb) {
        const int kt0 = kb * KB;

        // ---- S = Q K^T : four 16x16 score tiles (64 keys), reduction over d in 4 chunks
        v8f s[4];
#pragma unroll
        for (int st = 0; st < 4; ++st) s[st] = zero_v8f();
#pragma unroll
        for (int st = 0; st < 4; ++st) {
            const __bf16* krow = kbase + (long)(kt0 + st * 16 + lm) * HD;
#pragma unroll
            for (int c = 0; c < 4; ++c) {
                v16bf kf = combine16(*(const v8bf*)(krow + c * 32 + half * 8),
                                     *(const v8bf*)(krow + c * 32 + 16 + half * 8));
                s[st] = __builtin_amdgcn_wmma_f32_16x16x32_bf16(false, qf[c], false, kf,
                                                                (short)0, s[st], false, false);
            }
        }

        // ---- masked online softmax (fp32 per-row stats; shuffles stay in 16-lane halves)
        int keyv[4], pmv[4];
#pragma unroll
        for (int st = 0; st < 4; ++st) {
            keyv[st] = kt0 + st * 16 + lm;
            pmv[st]  = am[keyv[st]];
        }
        float p[4][8];
#pragma unroll
        for (int r = 0; r < 8; ++r) {
            const int qrow = m0 + r + 8 * half;
            float x[4];
#pragma unroll
            for (int st = 0; st < 4; ++st) {
                x[st] = s[st][r] * scale;
                if (keyv[st] > qrow || pmv[st] == 0) x[st] = -1e30f;
            }
            float mx = fmaxf(fmaxf(x[0], x[1]), fmaxf(x[2], x[3]));
            mx = fmaxf(mx, __shfl_xor(mx, 1));
            mx = fmaxf(mx, __shfl_xor(mx, 2));
            mx = fmaxf(mx, __shfl_xor(mx, 4));
            mx = fmaxf(mx, __shfl_xor(mx, 8));
            const float mnew = fmaxf(mrow[r], mx);
            const float sc = __expf(mrow[r] - mnew);
            float rs = 0.0f;
#pragma unroll
            for (int st = 0; st < 4; ++st) {
                const float e = __expf(x[st] - mnew);
                p[st][r] = e;
                rs += e;
            }
            rs += __shfl_xor(rs, 1);
            rs += __shfl_xor(rs, 2);
            rs += __shfl_xor(rs, 4);
            rs += __shfl_xor(rs, 8);
            lrow[r] = lrow[r] * sc + rs;
            mrow[r] = mnew;
#pragma unroll
            for (int nc = 0; nc < 8; ++nc) accO[nc][r] *= sc;
        }

        // ---- re-layout P (C-fragment) -> A-fragments via LDS (row-major 16x64 bf16)
#pragma unroll
        for (int r = 0; r < 8; ++r)
#pragma unroll
            for (int st = 0; st < 4; ++st)
                ldsP[(r + 8 * half) * KB + st * 16 + lm] = (__bf16)p[st][r];

        v16bf pa0 = combine16(*(const v8bf*)&ldsP[lm * KB + half * 8],
                              *(const v8bf*)&ldsP[lm * KB + 16 + half * 8]);
        v16bf pa1 = combine16(*(const v8bf*)&ldsP[lm * KB + 32 + half * 8],
                              *(const v8bf*)&ldsP[lm * KB + 48 + half * 8]);

        // ---- O += P @ V : 8 n-chunks of 16 over head_dim 128, two K=32 steps (64 keys)
#pragma unroll
        for (int nc = 0; nc < 8; ++nc) {
            const __bf16* vrow = vbase + (long)(nc * 16 + lm) * TT + kt0;
            v16bf vb0 = combine16(*(const v8bf*)(vrow + half * 8), *(const v8bf*)(vrow + 16 + half * 8));
            v16bf vb1 = combine16(*(const v8bf*)(vrow + 32 + half * 8), *(const v8bf*)(vrow + 48 + half * 8));
            accO[nc] = __builtin_amdgcn_wmma_f32_16x16x32_bf16(false, pa0, false, vb0,
                                                               (short)0, accO[nc], false, false);
            accO[nc] = __builtin_amdgcn_wmma_f32_16x16x32_bf16(false, pa1, false, vb1,
                                                               (short)0, accO[nc], false, false);
        }
    }

    // ---- normalize + store ctx as bf16 [B,T,C] at column h*HD + d
    __bf16* cb = ctx + (long)b * TT * CC + h * HD;
#pragma unroll
    for (int nc = 0; nc < 8; ++nc)
#pragma unroll
        for (int r = 0; r < 8; ++r) {
            const int qrow = m0 + r + 8 * half;
            const float o = accO[nc][r] / lrow[r];
            cb[(long)qrow * CC + nc * 16 + lm] = (__bf16)o;
        }
}

// ---------------------------------------------------------------- host side
extern "C" void kernel_launch(void* const* d_in, const int* in_sizes, int n_in,
                              void* d_out, int out_size, void* d_ws, size_t ws_size,
                              hipStream_t stream) {
    const float* x     = (const float*)d_in[0];
    const float* freqs = (const float*)d_in[1];
    const int*   amask = (const int*)d_in[2];
    const float* Wq    = (const float*)d_in[3];
    const float* Wk    = (const float*)d_in[4];
    const float* Wv    = (const float*)d_in[5];
    const float* Wo    = (const float*)d_in[6];
    float* out = (float*)d_out;

    char* ws = (char*)d_ws;
    size_t off = 0;
    auto alloc = [&](size_t bytes) -> void* {
        void* p = ws + off;
        off += (bytes + 255) & ~(size_t)255;
        return p;
    };

    __bf16* xb   = (__bf16*)alloc((size_t)MT * CC * 2);        // x bf16
    __bf16* wqb  = (__bf16*)alloc((size_t)CC * CC * 2);
    __bf16* wkb  = (__bf16*)alloc((size_t)KVC * CC * 2);
    __bf16* wvb  = (__bf16*)alloc((size_t)KVC * CC * 2);
    __bf16* wob  = (__bf16*)alloc((size_t)CC * CC * 2);
    float*  qf32 = (float*)alloc((size_t)MT * CC * 4);         // pre-RoPE Q
    float*  kf32 = (float*)alloc((size_t)MT * KVC * 4);        // pre-RoPE K
    float*  vf32 = (float*)alloc((size_t)MT * KVC * 4);
    __bf16* qrp  = (__bf16*)alloc((size_t)BB * NH * TT * HD * 2);   // [B,NH,T,HD]
    __bf16* krp  = (__bf16*)alloc((size_t)BB * NKV * TT * HD * 2);  // [B,NKV,T,HD]
    __bf16* vtp  = (__bf16*)alloc((size_t)BB * NKV * HD * TT * 2);  // [B,NKV,HD,T]
    __bf16* ctx  = (__bf16*)alloc((size_t)MT * CC * 2);        // attention output bf16

    auto cvt = [&](const float* s, __bf16* d, long n) {
        cvt_f32_bf16<<<dim3((unsigned)((n + 255) / 256)), dim3(256), 0, stream>>>(s, d, n);
    };
    // 1) downconvert activations + weights
    cvt(x,  xb,  (long)MT * CC);
    cvt(Wq, wqb, (long)CC * CC);
    cvt(Wk, wkb, (long)KVC * CC);
    cvt(Wv, wvb, (long)KVC * CC);
    cvt(Wo, wob, (long)CC * CC);

    // 2) QKV projections (WMMA GEMMs): O[m,n] = sum_k A[m,k]*W[n,k]
    gemm_bf16_nt<<<dim3(MT / 32, CC / 64),  dim3(32), 0, stream>>>(xb, wqb, qf32, MT, CC,  CC);
    gemm_bf16_nt<<<dim3(MT / 32, KVC / 64), dim3(32), 0, stream>>>(xb, wkb, kf32, MT, KVC, CC);
    gemm_bf16_nt<<<dim3(MT / 32, KVC / 64), dim3(32), 0, stream>>>(xb, wvb, vf32, MT, KVC, CC);

    // 3) RoPE + layout packs
    {
        long nq = (long)BB * NH * TT * (HD / 2);
        rope_q_kernel<<<dim3((unsigned)((nq + 255) / 256)), dim3(256), 0, stream>>>(qf32, freqs, qrp);
        long nk = (long)BB * NKV * TT * (HD / 2);
        rope_k_kernel<<<dim3((unsigned)((nk + 255) / 256)), dim3(256), 0, stream>>>(kf32, freqs, krp);
        long nv = (long)BB * NKV * TT * HD;
        pack_vT_kernel<<<dim3((unsigned)((nv + 255) / 256)), dim3(256), 0, stream>>>(vf32, vtp);
    }

    // 4) flash attention (WMMA for S and O, online softmax)
    flash_attn_kernel<<<dim3(TT / 16, NH, BB), dim3(32), 0, stream>>>(qrp, krp, vtp, amask, ctx);

    // 5) output projection -> fp32 d_out
    gemm_bf16_nt<<<dim3(MT / 32, CC / 64), dim3(32), 0, stream>>>(ctx, wob, out, MT, CC, CC);
}
